// LSTMmodel_28991029248533
// MI455X (gfx1250) — matrix operand order repaired and checked by reading, not compiled, not run on hardware
//
#include <hip/hip_runtime.h>
#include <hip/hip_bf16.h>
#include <math.h>
#include <stdint.h>

// ---------------------------------------------------------------------------
// Problem constants
// ---------------------------------------------------------------------------
#define BB   512
#define TT   64
#define DC   2048
#define DM   1024
#define FUS  3072          // DC + DM
#define HH   1024
#define G4   4096          // 4*H
#define NCLS 22
#define NPAD 32            // classifier N padded to 2 WMMA tiles
#define MTOT (BB*TT)       // 32768 rows of x / y

#define LDSROW 40          // halves per staged LDS row: 32 data + 8 pad (80B = 20 dw, conflict-free)
#define TILEH  (128*LDSROW) // halves per staged 128x32 tile (10240B)

typedef __attribute__((ext_vector_type(16))) _Float16 v16h;
typedef __attribute__((ext_vector_type(8)))  _Float16 v8h;
typedef __attribute__((ext_vector_type(8)))  float    v8f;

// ---------------------------------------------------------------------------
// WMMA fragment loaders (CDNA5 16x16x32 f16 layouts, ISA 7.12.2)
// ---------------------------------------------------------------------------
__device__ __forceinline__ v16h ld_fragA(const _Float16* __restrict__ p) {
  union { v16h v; v8h h[2]; } u;
  u.h[0] = *reinterpret_cast<const v8h*>(p);
  u.h[1] = *reinterpret_cast<const v8h*>(p + 16);
  return u.v;
}
__device__ __forceinline__ v16h ld_fragB(const _Float16* __restrict__ p) {
  union { v16h v; v8h h[2]; } u;
  u.h[0] = *reinterpret_cast<const v8h*>(p);
  u.h[1] = *reinterpret_cast<const v8h*>(p + 16);
  return u.v;
}
__device__ __forceinline__ v8f wmma_f16(v16h a, v16h b, v8f c) {
  return __builtin_amdgcn_wmma_f32_16x16x32_f16(false, a, false, b,
                                                (short)0, c, false, false);
}

// ---------------------------------------------------------------------------
// CDNA5 async global->LDS copy (ASYNCcnt-tracked) + waits, via inline asm.
// ---------------------------------------------------------------------------
__device__ __forceinline__ void async_cp16(void* lds, const void* g) {
  unsigned dst = (unsigned)(uintptr_t)lds;   // LDS offset within workgroup alloc
  asm volatile("global_load_async_to_lds_b128 %0, %1, off"
               :: "v"(dst), "v"(g) : "memory");
}
__device__ __forceinline__ void wait_async_le4() {
  asm volatile("s_wait_asynccnt 0x4" ::: "memory");
}
__device__ __forceinline__ void wait_async_le0() {
  asm volatile("s_wait_asynccnt 0x0" ::: "memory");
}

__device__ __forceinline__ float sigmoidf_fast(float x) {
  return 1.0f / (1.0f + __expf(-x));
}

// ---------------------------------------------------------------------------
// Prep kernels. NOTE: xh (and hence xproj) are stored T-MAJOR: row = t*B + b,
// so each step's xproj slice is one contiguous 8MB block.
// ---------------------------------------------------------------------------
__global__ void k_cvt_x(const float* __restrict__ cam,
                        const float* __restrict__ mot,
                        _Float16* __restrict__ xh) {
  long long idx = (long long)blockIdx.x * blockDim.x + threadIdx.x;
  if (idx >= (long long)MTOT * FUS) return;
  const int       k = (int)(idx % FUS);
  const long long m = idx / FUS;                 // t-major row: m = t*B + b
  const long long t = m / BB, b = m % BB;
  const long long src = b * TT + t;              // inputs are [B,T,*]
  float v = (k < DC) ? cam[src * DC + k] : mot[src * DM + (k - DC)];
  xh[idx] = (_Float16)v;
}

__global__ void k_cvt_w(const float* __restrict__ src,
                        _Float16* __restrict__ dst,
                        int rows_in, int rows_out, int cols) {
  long long idx = (long long)blockIdx.x * blockDim.x + threadIdx.x;
  if (idx >= (long long)rows_out * cols) return;
  int r = (int)(idx / cols);
  dst[idx] = (r < rows_in) ? (_Float16)src[idx] : (_Float16)0.0f;
}

__global__ void k_bias_sum(const float* __restrict__ bih,
                           const float* __restrict__ bhh,
                           float* __restrict__ bsum) {
  int i = blockIdx.x * blockDim.x + threadIdx.x;
  if (i < G4) bsum[i] = bih[i] + bhh[i];
}

__global__ void k_init_state(_Float16* __restrict__ h0,
                             float* __restrict__ c0) {
  int i = blockIdx.x * blockDim.x + threadIdx.x;
  if (i < BB * HH) { h0[i] = (_Float16)0.0f; c0[i] = 0.0f; }
}

// ---------------------------------------------------------------------------
// GEMM 1: xproj[32768,4096] = Xh[32768,3072] * Wih[4096,3072]^T + (bih+bhh)
// (rows t-major). 8 waves, WG tile 128x128, wave tile 32x64 (2x4 WMMA tiles).
// A/B 128x32-half tiles double-buffered in LDS via async copies.
// Steady-state loop body is branch-free; last K-iteration peeled.
// ---------------------------------------------------------------------------
__global__ __launch_bounds__(256) void k_gemm_xproj(
    const _Float16* __restrict__ A,   // [MTOT, FUS] t-major
    const _Float16* __restrict__ W,   // [G4, FUS]
    const float*    __restrict__ bias,
    float*          __restrict__ D) { // [MTOT, G4] t-major
  __shared__ _Float16 sA[2][TILEH];
  __shared__ _Float16 sB[2][TILEH];

  const int tid  = threadIdx.x;
  const int lane = tid & 31, wave = tid >> 5;
  const int l = lane & 15,   hi = lane >> 4;
  const int wm = wave & 3,   wn = wave >> 2;
  const int rowWG = blockIdx.y * 128;
  const int colWG = blockIdx.x * 128;

  v8f acc[2][4];
#pragma unroll
  for (int i = 0; i < 2; ++i)
#pragma unroll
    for (int j = 0; j < 4; ++j) acc[i][j] = (v8f)0.0f;

  // staging: 512 16B-chunks per matrix per K-slice; 2 A-chunks + 2 B-chunks/thread
  const int r0 = (tid * 2) >> 2, q0 = (tid * 2) & 3;   // chunk0; chunk1 = q0+1 (q0 is even)
  const _Float16* gA = A + (size_t)(rowWG + r0) * FUS + q0 * 8;
  const _Float16* gB = W + (size_t)(colWG + r0) * FUS + q0 * 8;
  const int ldsOff = r0 * LDSROW + q0 * 8;

  auto issue = [&](int bufi, int k0) {
    async_cp16(&sA[bufi][ldsOff],     gA + k0);
    async_cp16(&sA[bufi][ldsOff + 8], gA + k0 + 8);
    async_cp16(&sB[bufi][ldsOff],     gB + k0);
    async_cp16(&sB[bufi][ldsOff + 8], gB + k0 + 8);
  };
  auto compute = [&](int bufi) {
    v16h a[2], b[4];
#pragma unroll
    for (int i = 0; i < 2; ++i)
      a[i] = ld_fragA(&sA[bufi][(wm * 32 + 16 * i + l) * LDSROW + hi * 8]);
#pragma unroll
    for (int j = 0; j < 4; ++j)
      b[j] = ld_fragB(&sB[bufi][(wn * 64 + 16 * j + l) * LDSROW + hi * 8]);
#pragma unroll
    for (int i = 0; i < 2; ++i)
#pragma unroll
      for (int j = 0; j < 4; ++j)
        acc[i][j] = wmma_f16(a[i], b[j], acc[i][j]);
  };

  issue(0, 0);
  int buf = 0;
  for (int k0 = 0; k0 < FUS - 32; k0 += 32) {   // branch-free steady state
    issue(buf ^ 1, k0 + 32);
    wait_async_le4();
    __syncthreads();
    compute(buf);
    __syncthreads();
    buf ^= 1;
  }
  wait_async_le0();
  __syncthreads();
  compute(buf);

#pragma unroll
  for (int j = 0; j < 4; ++j) {
    const int   col = colWG + wn * 64 + 16 * j + l;
    const float bv  = bias[col];
#pragma unroll
    for (int i = 0; i < 2; ++i) {
      float* p = D + (size_t)(rowWG + wm * 32 + 16 * i + hi * 8) * G4 + col;
#pragma unroll
      for (int r = 0; r < 8; ++r) p[(size_t)r * G4] = acc[i][j][r] + bv;
    }
  }
}

// ---------------------------------------------------------------------------
// Fused LSTM step: each WG owns a 128-batch x 32-column block of h, computes
// the 4 gate tiles (cols g*H + c) seeded from t-major xproj, then applies the
// cell nonlinearity in registers. grid = (H/32 = 32, B/128 = 4)
// ---------------------------------------------------------------------------
__global__ __launch_bounds__(256) void k_step(
    const _Float16* __restrict__ Hprev,  // [BB, HH]
    const _Float16* __restrict__ W,      // [G4, HH] (gate-major rows)
    const float*    __restrict__ xproj,  // [MTOT, G4] t-major: row = t*B + b
    float*          __restrict__ C,      // [BB, HH]
    _Float16*       __restrict__ Hout,   // [BB, HH]
    int t) {
  __shared__ _Float16 sA[2][TILEH];
  __shared__ _Float16 sB[2][TILEH];

  const int tid  = threadIdx.x;
  const int lane = tid & 31, wave = tid >> 5;
  const int l = lane & 15,   hi = lane >> 4;
  const int wm = wave & 3,   wn = wave >> 2;
  const int rowWG = blockIdx.y * 128;          // batch base
  const int c0    = blockIdx.x * 32;           // column group within H
  const int colH  = c0 + wn * 16 + l;

  // seed accumulators from this step's contiguous xproj slice
  v8f acc[2][4];
  {
    const float* xp = xproj + ((size_t)t * BB + rowWG + wm * 32 + hi * 8) * G4 + colH;
#pragma unroll
    for (int i = 0; i < 2; ++i)
#pragma unroll
      for (int g = 0; g < 4; ++g)
#pragma unroll
        for (int r = 0; r < 8; ++r)
          acc[i][g][r] = xp[(size_t)(16 * i + r) * G4 + g * HH];
  }

  // staging: A rows = batch rows; B local row r -> gate g=r>>5, col j=r&31
  const int r0 = (tid * 2) >> 2, q0 = (tid * 2) & 3;
  const int gG = r0 >> 5, jG = r0 & 31;
  const _Float16* gA = Hprev + (size_t)(rowWG + r0) * HH + q0 * 8;
  const _Float16* gB = W + (size_t)(gG * HH + c0 + jG) * HH + q0 * 8;
  const int ldsOff = r0 * LDSROW + q0 * 8;

  auto issue = [&](int bufi, int k0) {
    async_cp16(&sA[bufi][ldsOff],     gA + k0);
    async_cp16(&sA[bufi][ldsOff + 8], gA + k0 + 8);
    async_cp16(&sB[bufi][ldsOff],     gB + k0);
    async_cp16(&sB[bufi][ldsOff + 8], gB + k0 + 8);
  };
  auto compute = [&](int bufi) {
    v16h a[2], b[4];
#pragma unroll
    for (int i = 0; i < 2; ++i)
      a[i] = ld_fragA(&sA[bufi][(wm * 32 + 16 * i + l) * LDSROW + hi * 8]);
#pragma unroll
    for (int g = 0; g < 4; ++g)
      b[g] = ld_fragB(&sB[bufi][(g * 32 + wn * 16 + l) * LDSROW + hi * 8]);
#pragma unroll
    for (int i = 0; i < 2; ++i)
#pragma unroll
      for (int g = 0; g < 4; ++g)
        acc[i][g] = wmma_f16(a[i], b[g], acc[i][g]);
  };

  issue(0, 0);
  int buf = 0;
  for (int k0 = 0; k0 < HH - 32; k0 += 32) {    // branch-free steady state
    issue(buf ^ 1, k0 + 32);
    wait_async_le4();
    __syncthreads();
    compute(buf);
    __syncthreads();
    buf ^= 1;
  }
  wait_async_le0();
  __syncthreads();
  compute(buf);

  // LSTM cell epilogue in registers (torch gate order i|f|g|o)
#pragma unroll
  for (int i = 0; i < 2; ++i) {
#pragma unroll
    for (int r = 0; r < 8; ++r) {
      const int b = rowWG + wm * 32 + 16 * i + hi * 8 + r;
      const size_t idx = (size_t)b * HH + colH;
      const float iv = sigmoidf_fast(acc[i][0][r]);
      const float fv = sigmoidf_fast(acc[i][1][r]);
      const float gv = tanhf(acc[i][2][r]);
      const float ov = sigmoidf_fast(acc[i][3][r]);
      const float cv = fv * C[idx] + iv * gv;
      C[idx] = cv;
      Hout[idx] = (_Float16)(ov * tanhf(cv));
    }
  }
}

// ---------------------------------------------------------------------------
// Classifier (batched over all steps): rows m = t*512 + b of Hall[32768,1024]
// out[(b*T+t)*22 + n] = Hall[m]*Wcls^T + bcls,  N padded to 32.
// ---------------------------------------------------------------------------
__global__ __launch_bounds__(256) void k_cls(
    const _Float16* __restrict__ Hall,   // [MTOT, HH] (slots 1..T)
    const _Float16* __restrict__ Wc,     // [NPAD, HH]
    const float*    __restrict__ bcls,   // [NCLS]
    float*          __restrict__ out) {  // [BB, TT, NCLS]
  const int tid  = threadIdx.x;
  const int lane = tid & 31, wave = tid >> 5;
  const int l = lane & 15,   hi = lane >> 4;
  const int rowBase = blockIdx.x * 256 + wave * 32;

  v8f acc[2][2];
#pragma unroll
  for (int i = 0; i < 2; ++i)
#pragma unroll
    for (int j = 0; j < 2; ++j) acc[i][j] = (v8f)0.0f;

  const _Float16* pa0 = Hall + (size_t)(rowBase + l) * HH + hi * 8;
  const _Float16* pb0 = Wc   + (size_t)l * HH + hi * 8;

  for (int k0 = 0; k0 < HH; k0 += 32) {
    v16h a[2], b[2];
#pragma unroll
    for (int i = 0; i < 2; ++i)
      a[i] = ld_fragA(pa0 + (size_t)(16 * i) * HH + k0);
#pragma unroll
    for (int j = 0; j < 2; ++j)
      b[j] = ld_fragB(pb0 + (size_t)(16 * j) * HH + k0);
#pragma unroll
    for (int i = 0; i < 2; ++i)
#pragma unroll
      for (int j = 0; j < 2; ++j)
        acc[i][j] = wmma_f16(a[i], b[j], acc[i][j]);
  }

#pragma unroll
  for (int j = 0; j < 2; ++j) {
    const int col = 16 * j + l;
    if (col < NCLS) {
      const float bv = bcls[col];
#pragma unroll
      for (int i = 0; i < 2; ++i) {
#pragma unroll
        for (int r = 0; r < 8; ++r) {
          const int m = rowBase + 16 * i + hi * 8 + r;
          const int t = m >> 9, b = m & (BB - 1);
          out[((size_t)b * TT + t) * NCLS + col] = acc[i][j][r] + bv;
        }
      }
    }
  }
}

// ---------------------------------------------------------------------------
// Host side
// ---------------------------------------------------------------------------
static inline size_t align_up(size_t v, size_t a) { return (v + a - 1) & ~(a - 1); }

extern "C" void kernel_launch(void* const* d_in, const int* in_sizes, int n_in,
                              void* d_out, int out_size, void* d_ws, size_t ws_size,
                              hipStream_t stream) {
  const float* cam  = (const float*)d_in[0];
  const float* mot  = (const float*)d_in[1];
  const float* Wih  = (const float*)d_in[2];
  const float* Whh  = (const float*)d_in[3];
  const float* bih  = (const float*)d_in[4];
  const float* bhh  = (const float*)d_in[5];
  const float* Wcls = (const float*)d_in[6];
  const float* bcls = (const float*)d_in[7];
  float* out = (float*)d_out;

  char* ws = (char*)d_ws;
  size_t off = 0;
  auto take = [&](size_t bytes) { void* p = ws + off; off = align_up(off + bytes, 256); return p; };

  _Float16* xh     = (_Float16*)take((size_t)MTOT * FUS * sizeof(_Float16)); // 201 MB (t-major)
  _Float16* Wih_h  = (_Float16*)take((size_t)G4 * FUS * sizeof(_Float16));   //  25 MB
  _Float16* Whh_h  = (_Float16*)take((size_t)G4 * HH * sizeof(_Float16));    //   8 MB
  _Float16* Wcls_h = (_Float16*)take((size_t)NPAD * HH * sizeof(_Float16));  //  64 KB
  float*    bias   = (float*)take((size_t)G4 * sizeof(float));
  float*    xproj  = (float*)take((size_t)MTOT * G4 * sizeof(float));        // 537 MB (t-major)
  float*    cbuf   = (float*)take((size_t)BB * HH * sizeof(float));          //   2 MB
  _Float16* hbuf   = (_Float16*)take((size_t)(TT + 1) * BB * HH * sizeof(_Float16)); // 68 MB
  (void)ws_size; (void)in_sizes; (void)n_in; (void)out_size;

  const size_t HSLOT = (size_t)BB * HH;

  // prep
  {
    long long n = (long long)MTOT * FUS;
    k_cvt_x<<<dim3((unsigned)((n + 255) / 256)), 256, 0, stream>>>(cam, mot, xh);
  }
  {
    long long n = (long long)G4 * FUS;
    k_cvt_w<<<dim3((unsigned)((n + 255) / 256)), 256, 0, stream>>>(Wih, Wih_h, G4, G4, FUS);
  }
  {
    long long n = (long long)G4 * HH;
    k_cvt_w<<<dim3((unsigned)((n + 255) / 256)), 256, 0, stream>>>(Whh, Whh_h, G4, G4, HH);
  }
  {
    long long n = (long long)NPAD * HH;
    k_cvt_w<<<dim3((unsigned)((n + 255) / 256)), 256, 0, stream>>>(Wcls, Wcls_h, NCLS, NPAD, HH);
  }
  k_bias_sum<<<dim3(G4 / 256), 256, 0, stream>>>(bih, bhh, bias);
  k_init_state<<<dim3((BB * HH) / 256), 256, 0, stream>>>(hbuf, cbuf);

  // big input-projection GEMM (async-LDS staged, WMMA)
  k_gemm_xproj<<<dim3(G4 / 128, MTOT / 128), 256, 0, stream>>>(xh, Wih_h, bias, xproj);

  // sequential recurrence: one fused GEMM+cell kernel per step
  for (int t = 0; t < TT; ++t) {
    k_step<<<dim3(HH / 32, BB / 128), 256, 0, stream>>>(
        hbuf + (size_t)t * HSLOT, Whh_h, xproj, cbuf,
        hbuf + (size_t)(t + 1) * HSLOT, t);
  }

  // batched classifier over all stored h_t (slots 1..T)
  k_cls<<<dim3(MTOT / 256), 256, 0, stream>>>(hbuf + HSLOT, Wcls_h, bcls, out);
}
